// WorkerModel_14388140441721
// MI455X (gfx1250) — matrix operand order repaired and checked by reading, not compiled, not run on hardware
//
#include <hip/hip_runtime.h>
#include <hip/hip_bf16.h>
#include <stdint.h>

#define BATCH 128
#define FDIM 128
#define NNODE 1025   // 32*32 grid cells + 1 meta node

typedef __attribute__((ext_vector_type(16))) __bf16   v16bf;
typedef __attribute__((ext_vector_type(8)))  float    v8f;
typedef __attribute__((ext_vector_type(8)))  uint32_t v8u;

union FragU { v8u u; v16bf b; };

__device__ __forceinline__ uint32_t pack_bf16x2(float a, float b) {
    uint32_t ua = __builtin_bit_cast(uint32_t, a);
    uint32_t ub = __builtin_bit_cast(uint32_t, b);
    ua = (ua + 0x7FFFu + ((ua >> 16) & 1u)) >> 16;   // RNE
    ub = (ub + 0x7FFFu + ((ub >> 16) & 1u)) >> 16;
    return (ua & 0xFFFFu) | (ub << 16);
}

__device__ __forceinline__ float elu1(float v) {
    return v > 0.f ? v : (__expf(v) - 1.f);
}

__device__ __forceinline__ void load_frag8(FragU& f, const uint32_t* p) {
    const uint4 b0 = *(const uint4*)(p);
    const uint4 b1 = *(const uint4*)(p + 4);
    f.u[0] = b0.x; f.u[1] = b0.y; f.u[2] = b0.z; f.u[3] = b0.w;
    f.u[4] = b1.x; f.u[5] = b1.y; f.u[6] = b1.z; f.u[7] = b1.w;
}

// ---------------------------------------------------------------- init x ----
__global__ void init_x_kernel(const float* __restrict__ gmap, float* __restrict__ x) {
    const long total = (long)BATCH * NNODE * FDIM;
    for (long i = (long)blockIdx.x * blockDim.x + threadIdx.x; i < total;
         i += (long)gridDim.x * blockDim.x) {
        const long b    = i / ((long)NNODE * FDIM);
        const long rem  = i - b * (long)NNODE * FDIM;
        const long node = rem / FDIM;
        const long f    = rem - node * FDIM;
        x[i] = (node < 1024) ? gmap[(b * 1024 + node) * FDIM + f] : 0.f;
    }
}

// ------------------------------------------------------------ weight pack ---
// W (K x N) fp32 -> fragment-major bf16 pairs:
//   Wf[frag*256 + lane*8 + j] = bf16x2{ W[k][n], W[k+1][n] }
//   frag = (k/32)*(Npad/16) + n/16 ; lane = (k%32>=16)*16 + n%16 ; j = (k%16)/2
// so a WMMA B fragment is 32 contiguous bytes per lane (2x b128 loads).
__global__ void pack_weight_kernel(const float* __restrict__ W, uint32_t* __restrict__ Wf,
                                   int K, int N, int Npad) {
    const int ntiles = Npad >> 4;
    const int total  = (K >> 1) * Npad;
    for (int i = blockIdx.x * blockDim.x + threadIdx.x; i < total;
         i += gridDim.x * blockDim.x) {
        const int frag = i >> 8;
        const int r    = i & 255;
        const int lane = r >> 3;
        const int j    = r & 7;
        const int kch  = frag / ntiles;
        const int nt   = frag - kch * ntiles;
        const int hi   = lane >> 4;
        const int ln   = lane & 15;
        const int k    = (kch << 5) + (hi << 4) + (j << 1);
        const int n    = (nt << 4) + ln;
        uint32_t v = 0;
        if (n < N)
            v = pack_bf16x2(W[(size_t)k * N + n], W[(size_t)(k + 1) * N + n]);
        Wf[i] = v;
    }
}

// ------------------------------------------------------------ meta reduce ---
// agg for the meta node: sum of all 1024 cell rows (8-way split + LDS reduce).
__global__ __launch_bounds__(1024) void meta_reduce_kernel(const float* __restrict__ x,
                                                           float* __restrict__ meta) {
    __shared__ float red[8][FDIM];
    const int b   = blockIdx.x;
    const int tid = threadIdx.x;
    const int f   = tid & 127;
    const int g   = tid >> 7;
    const float* xb = x + (size_t)b * NNODE * FDIM + f;
    float s = 0.f;
    const int c0 = g * 128;
    for (int cell = 0; cell < 128; ++cell) s += xb[(size_t)(c0 + cell) * FDIM];
    red[g][f] = s;
    __syncthreads();
    if (g == 0) {
        float t = red[0][f];
        #pragma unroll
        for (int gg = 1; gg < 8; ++gg) t += red[gg][f];
        meta[b * FDIM + f] = t;
    }
}

// --------------------------------------------------------------- GNN layer --
// out = elu(x@Ws + agg@Wn + bias); aggregation fused into bf16 LDS staging,
// weights staged per-phase in LDS in fragment order. All 8 B fragments of a
// k-step are preloaded before the 8 back-to-back WMMAs.
// grid: (9 row-tiles, 128 batches); block: 256 threads (8 waves).

#define WMMA_STEP(nt, RA)                                                       \
    acc[nt] = __builtin_amdgcn_wmma_f32_16x16x32_bf16(                          \
        false, a.b, false, bbs[nt].b, (short)0, acc[nt], RA, false);

__global__ __launch_bounds__(256) void gnn_layer_kernel(
    const float* __restrict__ x, const float* __restrict__ meta_agg,
    const uint32_t* __restrict__ WsF, const uint32_t* __restrict__ WnF,
    const float* __restrict__ bias, float* __restrict__ xout)
{
    __shared__ uint32_t xs[128][64];    // x   tile, bf16 pairs (32 KB)
    __shared__ uint32_t gs[128][64];    // agg tile, bf16 pairs (32 KB)
    __shared__ uint32_t wsh[32][256];   // current phase's weight fragments (32 KB)

    const int b    = blockIdx.y;
    const int row0 = blockIdx.x * 128;
    const int tid  = threadIdx.x;
    const float* xb    = x + (size_t)b * NNODE * FDIM;
    const float* xmeta = xb + 1024 * FDIM;

    // ---- stage A tiles: x rows + fused neighbor aggregation, float4-wide ----
    for (int i = tid; i < 128 * 32; i += 256) {
        const int row  = i >> 5;
        const int q    = i & 31;
        const int f0   = q << 2;
        const int node = row0 + row;
        uint32_t xv0 = 0, xv1 = 0, gv0 = 0, gv1 = 0;
        if (node < NNODE) {
            const float4 xr = *(const float4*)(xb + (size_t)node * FDIM + f0);
            xv0 = pack_bf16x2(xr.x, xr.y);
            xv1 = pack_bf16x2(xr.z, xr.w);
            float4 s;
            if (node == 1024) {                       // meta node: precomputed sum
                s = *(const float4*)(meta_agg + b * FDIM + f0);
            } else {                                  // grid cell: 4-neigh + meta
                const int r = node >> 5, c = node & 31;
                s = *(const float4*)(xmeta + f0);
                if (r > 0)  { float4 t = *(const float4*)(xb + (size_t)(node - 32) * FDIM + f0); s.x += t.x; s.y += t.y; s.z += t.z; s.w += t.w; }
                if (r < 31) { float4 t = *(const float4*)(xb + (size_t)(node + 32) * FDIM + f0); s.x += t.x; s.y += t.y; s.z += t.z; s.w += t.w; }
                if (c > 0)  { float4 t = *(const float4*)(xb + (size_t)(node -  1) * FDIM + f0); s.x += t.x; s.y += t.y; s.z += t.z; s.w += t.w; }
                if (c < 31) { float4 t = *(const float4*)(xb + (size_t)(node +  1) * FDIM + f0); s.x += t.x; s.y += t.y; s.z += t.z; s.w += t.w; }
            }
            gv0 = pack_bf16x2(s.x, s.y);
            gv1 = pack_bf16x2(s.z, s.w);
        }
        xs[row][q * 2]     = xv0;
        xs[row][q * 2 + 1] = xv1;
        gs[row][q * 2]     = gv0;
        gs[row][q * 2 + 1] = gv1;
    }
    __syncthreads();

    const int wave = tid >> 5;
    const int lane = tid & 31;
    const int hi   = lane >> 4;
    const int ln   = lane & 15;
    const int arow = (wave << 4) + ln;

    v8f acc[8];
    const v8f zero = {0.f, 0.f, 0.f, 0.f, 0.f, 0.f, 0.f, 0.f};
    #pragma unroll
    for (int t = 0; t < 8; ++t) acc[t] = zero;

    #pragma unroll
    for (int phase = 0; phase < 2; ++phase) {
        // ---- stage this phase's 32 KB of weight fragments into LDS ----
        {
            const uint4* src = (const uint4*)(phase ? WnF : WsF);
            uint4* dst = (uint4*)&wsh[0][0];
            for (int i = tid; i < 2048; i += 256) dst[i] = src[i];
        }
        __syncthreads();

        const uint32_t (*As)[64] = phase ? gs : xs;
        #pragma unroll
        for (int kk = 0; kk < 4; ++kk) {            // K = 128 = 4 x 32
            const int pb = kk * 16;
            FragU a;
            #pragma unroll
            for (int j = 0; j < 8; ++j)             // A 16x32 bf16 layout
                a.u[j] = As[arow][pb + j + (hi ? 8 : 0)];
            FragU bbs[8];
            #pragma unroll
            for (int nt = 0; nt < 8; ++nt)          // preload all B fragments
                load_frag8(bbs[nt], &wsh[kk * 8 + nt][lane * 8]);
            WMMA_STEP(0, true)  WMMA_STEP(1, true)  WMMA_STEP(2, true)
            WMMA_STEP(3, true)  WMMA_STEP(4, true)  WMMA_STEP(5, true)
            WMMA_STEP(6, true)  WMMA_STEP(7, false)
        }
        __syncthreads();   // wsh reused by next phase
    }

    // ---- epilogue: bias + ELU, fp32 store ----
    float* ob = xout + (size_t)b * NNODE * FDIM;
    #pragma unroll
    for (int nt = 0; nt < 8; ++nt) {
        const int col = nt * 16 + ln;
        const float bv = bias[col];
        #pragma unroll
        for (int v = 0; v < 8; ++v) {
            const int node = row0 + (wave << 4) + v + (hi ? 8 : 0);
            if (node < NNODE)
                ob[(size_t)node * FDIM + col] = elu1(acc[nt][v] + bv);
        }
    }
}

// -------------------------------------------------------------- gather ------
// Replicates the reference's padded-map gather (index = (r+dr+1)*32 + (c+dc+1)
// into a 34-wide padded map).
__global__ __launch_bounds__(128) void gather_state_kernel(
    const float* __restrict__ x, const int* __restrict__ pos, float* __restrict__ state)
{
    const int b = blockIdx.x;
    const int f = threadIdx.x;
    const int r = pos[b * 2 + 0];
    const int c = pos[b * 2 + 1];
    const int offr[5] = {-1, 0, 1, 0, 0};
    const int offc[5] = { 0,-1, 0, 1, 0};
    #pragma unroll
    for (int o = 0; o < 5; ++o) {
        const int pidx = (r + offr[o] + 1) * 32 + (c + offc[o] + 1);
        const int pr = pidx / 34, pc = pidx % 34;
        float v = 0.f;
        if (pr >= 1 && pr <= 32 && pc >= 1 && pc <= 32)
            v = x[((size_t)b * NNODE + (pr - 1) * 32 + (pc - 1)) * FDIM + f];
        state[(size_t)b * 640 + o * FDIM + f] = v;
    }
}

// ------------------------------------------------------------- MLP GEMM -----
// out[M x Nout] = act(A[M x K] @ W[K x Npad] + bias) (+ log-mask on last layer)
// grid: (Npad/128, ceil(M/128)); block: 256 threads (8 waves).
__global__ __launch_bounds__(256) void mlp_gemm_kernel(
    const float* __restrict__ A, const uint32_t* __restrict__ Wp,
    const float* __restrict__ bias, float* __restrict__ out,
    int M, int K, int Nout, int applyElu, const int* __restrict__ mask, int maskN)
{
    __shared__ uint32_t As[128][16];   // one 128x32 bf16 A chunk (8 KB)

    const int Npad   = gridDim.x * 128;
    const int nblk   = blockIdx.x * 128;
    const int ntile0 = nblk >> 4;
    const int row0   = blockIdx.y * 128;
    const int tid    = threadIdx.x;
    const int wave   = tid >> 5;
    const int lane   = tid & 31;
    const int hi     = lane >> 4;
    const int ln     = lane & 15;
    const int arow   = (wave << 4) + ln;

    v8f acc[8];
    const v8f zero = {0.f, 0.f, 0.f, 0.f, 0.f, 0.f, 0.f, 0.f};
    #pragma unroll
    for (int t = 0; t < 8; ++t) acc[t] = zero;

    for (int kc = 0; kc < K; kc += 32) {
        if (kc + 32 < K)  // hint next A chunk into cache (global_prefetch_b8)
            __builtin_prefetch(A + (size_t)(row0 + (tid >> 1)) * K + kc + 32, 0, 1);

        for (int i = tid; i < 128 * 16; i += 256) {
            const int row  = i >> 4;
            const int p    = i & 15;
            const int grow = row0 + row;
            uint32_t v = 0;
            if (grow < M) {
                const float2 t = *(const float2*)(A + (size_t)grow * K + kc + p * 2);
                v = pack_bf16x2(t.x, t.y);
            }
            As[row][p] = v;
        }
        __syncthreads();

        FragU a;
        #pragma unroll
        for (int j = 0; j < 8; ++j)
            a.u[j] = As[arow][j + (hi ? 8 : 0)];
        const int fragRow = (kc >> 5) * (Npad >> 4);
        FragU bbs[8];
        #pragma unroll
        for (int nt = 0; nt < 8; ++nt)          // preload all B fragments
            load_frag8(bbs[nt],
                       Wp + (((size_t)fragRow + ntile0 + nt) << 8) + lane * 8);
        WMMA_STEP(0, true)  WMMA_STEP(1, true)  WMMA_STEP(2, true)
        WMMA_STEP(3, true)  WMMA_STEP(4, true)  WMMA_STEP(5, true)
        WMMA_STEP(6, true)  WMMA_STEP(7, false)
        __syncthreads();
    }

    #pragma unroll
    for (int nt = 0; nt < 8; ++nt) {
        const int col = nblk + nt * 16 + ln;
        if (col >= Nout) continue;
        const float bv = bias[col];
        #pragma unroll
        for (int v = 0; v < 8; ++v) {
            const int row = row0 + (wave << 4) + v + (hi ? 8 : 0);
            if (row >= M) continue;
            float val = acc[nt][v] + bv;
            if (applyElu) val = elu1(val);
            if (mask) val += mask[(size_t)row * maskN + col] ? 0.f : -3.402823466e38f;
            out[(size_t)row * Nout + col] = val;
        }
    }
}

// ----------------------------------------------------------------- launch ---
extern "C" void kernel_launch(void* const* d_in, const int* in_sizes, int n_in,
                              void* d_out, int out_size, void* d_ws, size_t ws_size,
                              hipStream_t stream) {
    (void)in_sizes; (void)n_in; (void)out_size; (void)ws_size;

    const float* gmap  = (const float*)d_in[0];
    const int*   pos   = (const int*)d_in[1];
    const int*   amask = (const int*)d_in[2];
    const float* Ws    = (const float*)d_in[3];
    const float* Wn    = (const float*)d_in[4];
    const float* bs    = (const float*)d_in[5];
    const float* W_d1  = (const float*)d_in[6];
    const float* b_d1  = (const float*)d_in[7];
    const float* W_d2  = (const float*)d_in[8];
    const float* b_d2  = (const float*)d_in[9];
    const float* W_d3  = (const float*)d_in[10];
    const float* b_d3  = (const float*)d_in[11];
    const float* W_p1  = (const float*)d_in[12];
    const float* b_p1  = (const float*)d_in[13];
    const float* W_p2  = (const float*)d_in[14];
    const float* b_p2  = (const float*)d_in[15];
    const float* W_p3  = (const float*)d_in[16];
    const float* b_p3  = (const float*)d_in[17];

    char*  ws  = (char*)d_ws;
    size_t off = 0;
    auto take = [&](size_t bytes) -> char* {
        char* p = ws + off;
        off += (bytes + 255) & ~(size_t)255;
        return p;
    };

    const size_t xbytes = (size_t)BATCH * NNODE * FDIM * 4;
    float*    xA    = (float*)take(xbytes);
    float*    xB    = (float*)take(xbytes);
    float*    meta  = (float*)take((size_t)BATCH * FDIM * 4);
    uint32_t* WsP   = (uint32_t*)take(3 * 64 * 128 * 4);
    uint32_t* WnP   = (uint32_t*)take(3 * 64 * 128 * 4);
    uint32_t* Wd1P  = (uint32_t*)take(320 * 512 * 4);
    uint32_t* Wd2P  = (uint32_t*)take(256 * 512 * 4);
    uint32_t* Wd3P  = (uint32_t*)take(256 * 256 * 4);
    uint32_t* Wp1P  = (uint32_t*)take(128 * 256 * 4);
    uint32_t* Wp2P  = (uint32_t*)take(128 * 256 * 4);
    uint32_t* Wp3P  = (uint32_t*)take(128 * 128 * 4);
    float*    state = (float*)take(128 * 640 * 4);
    float*    h1    = (float*)take(128 * 512 * 4);
    float*    h2    = (float*)take(128 * 512 * 4);
    float*    feat  = (float*)take(128 * 256 * 4);
    float*    p1b   = (float*)take(128 * 256 * 4);
    float*    p2b   = (float*)take(128 * 256 * 4);

    // init x (cells from gmap, meta row zero)
    init_x_kernel<<<8192, 256, 0, stream>>>(gmap, xA);

    // pack all weights into fragment-major bf16 layout
    for (int l = 0; l < 3; ++l) {
        pack_weight_kernel<<<32, 256, 0, stream>>>(Ws + l * 128 * 128, WsP + l * 64 * 128, 128, 128, 128);
        pack_weight_kernel<<<32, 256, 0, stream>>>(Wn + l * 128 * 128, WnP + l * 64 * 128, 128, 128, 128);
    }
    pack_weight_kernel<<<640, 256, 0, stream>>>(W_d1, Wd1P, 640, 512, 512);
    pack_weight_kernel<<<512, 256, 0, stream>>>(W_d2, Wd2P, 512, 512, 512);
    pack_weight_kernel<<<256, 256, 0, stream>>>(W_d3, Wd3P, 512, 256, 256);
    pack_weight_kernel<<<128, 256, 0, stream>>>(W_p1, Wp1P, 256, 256, 256);
    pack_weight_kernel<<<128, 256, 0, stream>>>(W_p2, Wp2P, 256, 256, 256);
    pack_weight_kernel<<< 64, 256, 0, stream>>>(W_p3, Wp3P, 256,  19, 128);

    // 3 GNN layers (ping-pong)
    float* cur = xA;
    float* nxt = xB;
    for (int l = 0; l < 3; ++l) {
        meta_reduce_kernel<<<BATCH, 1024, 0, stream>>>(cur, meta);
        gnn_layer_kernel<<<dim3(9, BATCH), 256, 0, stream>>>(
            cur, meta, WsP + l * 64 * 128, WnP + l * 64 * 128, bs + l * 128, nxt);
        float* t = cur; cur = nxt; nxt = t;
    }

    // gather 5-neighborhood state
    gather_state_kernel<<<BATCH, 128, 0, stream>>>(cur, pos, state);

    // MLP head
    mlp_gemm_kernel<<<dim3(4, 1), 256, 0, stream>>>(state, Wd1P, b_d1, h1,   128, 640, 512, 1, nullptr, 0);
    mlp_gemm_kernel<<<dim3(4, 1), 256, 0, stream>>>(h1,    Wd2P, b_d2, h2,   128, 512, 512, 1, nullptr, 0);
    mlp_gemm_kernel<<<dim3(2, 1), 256, 0, stream>>>(h2,    Wd3P, b_d3, feat, 128, 512, 256, 1, nullptr, 0);
    mlp_gemm_kernel<<<dim3(2, 1), 256, 0, stream>>>(feat,  Wp1P, b_p1, p1b,  128, 256, 256, 1, nullptr, 0);
    mlp_gemm_kernel<<<dim3(2, 1), 256, 0, stream>>>(p1b,   Wp2P, b_p2, p2b,  128, 256, 256, 1, nullptr, 0);
    mlp_gemm_kernel<<<dim3(1, 1), 256, 0, stream>>>(p2b,   Wp3P, b_p3, (float*)d_out,
                                                    128, 256, 19, 0, amask, 19);
}
